// Pruno2DMulti_44083544326640
// MI455X (gfx1250) — compile-verified, matrix operand order checked by reading, not run on hardware
//
#include <hip/hip_runtime.h>
#include <hip/hip_bf16.h>

// Problem constants (from reference): x is (B, S, C) float32
#define B_   32
#define S_   3136          // 49 * 64 : K=64 divides exactly
#define C_   256
#define KSTEP 64
#define THRESH 1568        // SIMILARITY(0.5) * S : percent < 0.5  <=>  count < 1568

typedef __attribute__((ext_vector_type(8))) int v8i;

// ---------------- workspace layout (bytes) ----------------
//  mean   : [0,            32768)   B*C float
//  maskf  : [32768,        65536)   B*C float
//  G      : [65536,      8454144)   B*C*C int
//  packed : [8454144,   34144256)   B*C*S u8   (live^T, row = channel, K contiguous)
//  psum   : [34144256,  35192832)   B*32*C float  (also absorbs prefetch overrun)
#define WS_MEAN   0
#define WS_MASK   32768
#define WS_G      65536
#define WS_PACKED 8454144
#define WS_PSUM   34144256

// ---- 1) deterministic partial column sums over S (32 chunks of 98 rows) ----
__global__ void psum_kernel(const float* __restrict__ x, float* __restrict__ psum) {
    const int chunk = blockIdx.x;          // 0..31
    const int b     = blockIdx.y;          // 0..B-1
    const int c     = threadIdx.x;         // 0..255
    const float* p = x + (size_t)b * S_ * C_ + (size_t)chunk * 98 * C_ + c;
    float s = 0.0f;
    #pragma unroll 7
    for (int r = 0; r < 98; ++r) s += p[(size_t)r * C_];
    psum[((size_t)b * 32 + chunk) * C_ + c] = s;
}

// ---- 2) reduce partials -> per-channel mean ----
__global__ void mean_kernel(const float* __restrict__ psum, float* __restrict__ mean) {
    const int b = blockIdx.x;
    const int c = threadIdx.x;
    float s = 0.0f;
    #pragma unroll
    for (int k = 0; k < 32; ++k) s += psum[((size_t)b * 32 + k) * C_ + c];
    mean[(size_t)b * C_ + c] = s * (1.0f / (float)S_);
}

// ---- 3) threshold + LDS-tiled transpose: live(B,S,C) -> packed u8 (B,C,S) ----
__global__ void pack_kernel(const float* __restrict__ x, const float* __restrict__ mean,
                            unsigned char* __restrict__ packed) {
    __shared__ unsigned char tile[32][33];          // +1 to dodge bank conflicts
    const int s0 = blockIdx.x * 32;                 // 98 tiles over S
    const int c0 = blockIdx.y * 32;                 // 8 tiles over C
    const int b  = blockIdx.z;
    const int tx = threadIdx.x;                     // 0..31
    const int ty = threadIdx.y;                     // 0..7
    const float mv = mean[(size_t)b * C_ + c0 + tx];   // channel = c0+tx during reads
    #pragma unroll
    for (int r = 0; r < 4; ++r) {
        const int srow = ty * 4 + r;
        const float v = x[(size_t)b * S_ * C_ + (size_t)(s0 + srow) * C_ + c0 + tx];
        tile[srow][tx] = (v > mv) ? (unsigned char)1 : (unsigned char)0;
    }
    __syncthreads();
    #pragma unroll
    for (int r = 0; r < 4; ++r) {
        const int crow = ty * 4 + r;                // channel-local row on write
        packed[(size_t)b * C_ * S_ + (size_t)(c0 + crow) * S_ + s0 + tx] = tile[tx][crow];
    }
}

// Build an 8-bit 16x64 WMMA fragment for this lane.
// ISA layout (CDNA5 ISA): lane m=lane&15, half=lane>>4;
// qwords come from row m at byte offsets k + 8*half + {0,16,32,48}.
__device__ __forceinline__ v8i make_frag8(const unsigned char* p) {
    const unsigned long long* q = (const unsigned long long*)p;
    const unsigned long long q0 = q[0], q1 = q[2], q2 = q[4], q3 = q[6];
    v8i f;
    f[0] = (int)q0; f[1] = (int)(q0 >> 32);
    f[2] = (int)q1; f[3] = (int)(q1 >> 32);
    f[4] = (int)q2; f[5] = (int)(q2 >> 32);
    f[6] = (int)q3; f[7] = (int)(q3 >> 32);
    return f;
}

// ---- 4) Gram matrix G = M * M^T (M = packed live^T, C x S, u8 0/1) via IU8 WMMA ----
// grid (16, 2, B), block 128 (4 waves). One wave -> TWO 16x16 tiles (ti,tj) and
// (ti,tj+8) sharing the A fragment: 12 b64 loads -> 2 independent WMMAs per k-step
// (independent accumulators = no accumulator ping-pong moves between WMMAs).
__global__ void gram_kernel(const unsigned char* __restrict__ packed, int* __restrict__ G) {
    const int b    = blockIdx.z;
    const int ti   = blockIdx.x;                          // output tile row 0..15
    const int tj0  = blockIdx.y * 4 + (threadIdx.x >> 5); // 0..7
    const int tj1  = tj0 + 8;                             // 8..15
    const int lane = threadIdx.x & 31;
    const int n    = lane & 15;
    const int half = lane >> 4;

    const unsigned char* baseM = packed + (size_t)b * C_ * S_;
    const unsigned char* pA  = baseM + (size_t)(ti  * 16 + n) * S_ + 8 * half;
    const unsigned char* pB0 = baseM + (size_t)(tj0 * 16 + n) * S_ + 8 * half;
    const unsigned char* pB1 = baseM + (size_t)(tj1 * 16 + n) * S_ + 8 * half;

    v8i acc0 = {0, 0, 0, 0, 0, 0, 0, 0};
    v8i acc1 = {0, 0, 0, 0, 0, 0, 0, 0};
    for (int k = 0; k < S_; k += KSTEP) {
        // Unguarded stream-ahead prefetch (worst case lands in the psum region
        // that follows `packed` inside the same workspace allocation).
        // locality=3 -> near-scope prefetch so the line lands in the WGP cache.
        __builtin_prefetch(pA  + k + 2048, 0, 3);
        __builtin_prefetch(pB0 + k + 2048, 0, 3);
        __builtin_prefetch(pB1 + k + 2048, 0, 3);
        const v8i a  = make_frag8(pA  + k);
        const v8i b0 = make_frag8(pB0 + k);
        const v8i b1 = make_frag8(pB1 + k);
        // unsigned x unsigned 0/1 bytes -> exact co-live counts in i32
        acc0 = __builtin_amdgcn_wmma_i32_16x16x64_iu8(false, a, false, b0, acc0, false, false);
        acc1 = __builtin_amdgcn_wmma_i32_16x16x64_iu8(false, a, false, b1, acc1, false, false);
    }

    // C/D layout: lanes 0-15 -> M = v, lanes 16-31 -> M = v+8; N = lane&15
    int* g0 = G + (size_t)b * C_ * C_ + (size_t)(ti * 16) * C_ + tj0 * 16;
    int* g1 = G + (size_t)b * C_ * C_ + (size_t)(ti * 16) * C_ + tj1 * 16;
    #pragma unroll
    for (int v = 0; v < 8; ++v) {
        const int row = v + 8 * half;
        g0[(size_t)row * C_ + n] = acc0[v];
        g1[(size_t)row * C_ + n] = acc1[v];
    }
}

// ---- 5) per-channel survival: max off-diagonal count < 1568 ----
__global__ void rowmask_kernel(const int* __restrict__ G, float* __restrict__ maskf) {
    const int b = blockIdx.x;
    const int i = threadIdx.x;
    const int* row = G + (size_t)b * C_ * C_ + (size_t)i * C_;
    int m = -1;
    for (int j = 0; j < C_; ++j) {
        if (j == i) continue;                         // diagonal forced to "pass"
        const int v = row[j];
        m = (v > m) ? v : m;
    }
    maskf[(size_t)b * C_ + i] = (m < THRESH) ? 1.0f : 0.0f;
}

// ---- 6) out = x * mask (float4 vectorized; C%4==0 keeps mask loads aligned) ----
__global__ void apply_kernel(const float* __restrict__ x, const float* __restrict__ maskf,
                             float* __restrict__ out) {
    const size_t t = (size_t)blockIdx.x * blockDim.x + threadIdx.x;
    const size_t e = t * 4;                           // element index, 16B aligned
    const int c = (int)(e & (C_ - 1));
    const int b = (int)(e / ((size_t)S_ * C_));
    const float4 xv = *(const float4*)(x + e);
    const float4 mv = *(const float4*)(maskf + (size_t)b * C_ + c);
    float4 o;
    o.x = xv.x * mv.x; o.y = xv.y * mv.y; o.z = xv.z * mv.z; o.w = xv.w * mv.w;
    *(float4*)(out + e) = o;
}

extern "C" void kernel_launch(void* const* d_in, const int* in_sizes, int n_in,
                              void* d_out, int out_size, void* d_ws, size_t ws_size,
                              hipStream_t stream) {
    (void)in_sizes; (void)n_in; (void)out_size; (void)ws_size;
    const float* x = (const float*)d_in[0];
    float* out = (float*)d_out;

    char* ws = (char*)d_ws;
    float*         mean   = (float*)(ws + WS_MEAN);
    float*         maskf  = (float*)(ws + WS_MASK);
    int*           G      = (int*)(ws + WS_G);
    unsigned char* packed = (unsigned char*)(ws + WS_PACKED);
    float*         psum   = (float*)(ws + WS_PSUM);

    // 1) partial column sums (deterministic fixed-order reduction)
    psum_kernel<<<dim3(32, B_), 256, 0, stream>>>(x, psum);
    // 2) mean
    mean_kernel<<<dim3(B_), 256, 0, stream>>>(psum, mean);
    // 3) threshold + transpose-pack to u8 (B, C, S)
    pack_kernel<<<dim3(S_ / 32, C_ / 32, B_), dim3(32, 8), 0, stream>>>(x, mean, packed);
    // 4) IU8 WMMA Gram matrix (packed buffer is L2-resident: 25.7 MB << 192 MB)
    gram_kernel<<<dim3(C_ / 16, C_ / 128, B_), 128, 0, stream>>>(packed, G);
    // 5) row max vs threshold -> float mask
    rowmask_kernel<<<dim3(B_), C_, 0, stream>>>(G, maskf);
    // 6) broadcast multiply
    const size_t total4 = (size_t)B_ * S_ * C_ / 4;   // 6,422,528 -> exactly 25088 blocks
    apply_kernel<<<dim3((unsigned)(total4 / 256)), 256, 0, stream>>>(x, maskf, out);
}
